// AttentionModule_38104949850191
// MI455X (gfx1250) — matrix operand order repaired and checked by reading, not compiled, not run on hardware
//
#include <hip/hip_runtime.h>

typedef _Float16 v16h __attribute__((ext_vector_type(16)));
typedef float    v8f  __attribute__((ext_vector_type(8)));
typedef _Float16 h2   __attribute__((ext_vector_type(2)));

#define DEVI __device__ __forceinline__

constexpr int Bn = 4;      // batch
constexpr int C  = 64;     // channels / head dim
constexpr int L  = 4096;   // sequence length
constexpr int LDSS = 72;   // LDS row stride in f16 (64+8 pad; 144 B = 9*16 B)

// ---------------------------------------------------------------------------
// WMMA helper: D = A(16x32 f16) * B(32x16 f16) + C(16x16 f32)
// ---------------------------------------------------------------------------
DEVI v8f wmma_f16(v16h a, v16h b, v8f c) {
  return __builtin_amdgcn_wmma_f32_16x16x32_f16(
      /*neg_a=*/false, a, /*neg_b=*/false, b,
      /*c_mod=*/(short)0, c, /*reuse_a=*/false, /*reuse_b=*/false);
}

// A-matrix 16x32 f16 fragment (CDNA5 layout):
//   lane l (row m = l&15), VGPR v: K = (v<4?0:16) + 2*(v&3) + (l>=16 ? 8 : 0)
DEVI v16h load_A(const _Float16* base, int m, int hi, int koff) {
  v16h a;
#pragma unroll
  for (int v = 0; v < 8; ++v) {
    int K = koff + ((v >> 2) << 4) + ((v & 3) << 1) + (hi << 3);
    h2 p = *reinterpret_cast<const h2*>(base + m * LDSS + K);
    a[2 * v]     = p[0];
    a[2 * v + 1] = p[1];
  }
  return a;
}

// B-matrix 32x16 f16 fragment (CDNA5 layout):
//   lane l (col n = l&15), VGPR v: K = (v<4?0:16) + 2*(v&3) + (l>=16 ? 8 : 0)
DEVI v16h load_B(const _Float16* base, int n, int hi, int koff) {
  v16h b;
#pragma unroll
  for (int v = 0; v < 8; ++v) {
    int K = koff + ((v >> 2) << 4) + ((v & 3) << 1) + (hi << 3);
    h2 p = *reinterpret_cast<const h2*>(base + n * LDSS + K);
    b[2 * v]     = p[0];
    b[2 * v + 1] = p[1];
  }
  return b;
}

// ---------------------------------------------------------------------------
// CDNA5 async global->LDS (ASYNCcnt-tracked, no VGPR round trip).
//   global_load_async_to_lds_b128 vdst(lds byte addr), vaddr(64b), off
// ---------------------------------------------------------------------------
DEVI void async_ld_b128(unsigned lds_byte_addr, const void* gaddr) {
  asm volatile("global_load_async_to_lds_b128 %0, %1, off"
               :
               : "v"(lds_byte_addr), "v"((unsigned long long)(uintptr_t)gaddr)
               : "memory");
}

template <int N>
DEVI void wait_asynccnt() {
#if __has_builtin(__builtin_amdgcn_s_wait_asynccnt)
  __builtin_amdgcn_s_wait_asynccnt(N);
#else
  asm volatile("s_wait_asynccnt %0" ::"n"(N) : "memory");
#endif
}

// ---------------------------------------------------------------------------
// Kernel 1: QKV projection (kernel-size-1 conv == per-position linear).
//   q[b,l,o] = sum_c wq[o,c]*x[b,c,l] + bq[o]   (f16, [B,L,64])
//   k       : same layout
//   v       : transposed [B,64,L] so P*V's B-operand has contiguous K-pairs.
// ---------------------------------------------------------------------------
__global__ __launch_bounds__(256) void proj_qkv(
    const float* __restrict__ x,
    const float* __restrict__ wq, const float* __restrict__ bq,
    const float* __restrict__ wk, const float* __restrict__ bk,
    const float* __restrict__ wv, const float* __restrict__ bv,
    _Float16* __restrict__ q16, _Float16* __restrict__ k16,
    _Float16* __restrict__ v16t) {
  const int bid = blockIdx.x;
  const int lt  = bid & 15;
  const int o   = (bid >> 4) & 63;
  const int b   = bid >> 10;
  const int l   = lt * 256 + threadIdx.x;

  float aq = bq[o], ak = bk[o], av = bv[o];
  const float* xb  = x + (size_t)b * C * L + l;
  const float* wqr = wq + o * C;
  const float* wkr = wk + o * C;
  const float* wvr = wv + o * C;
#pragma unroll 8
  for (int c = 0; c < C; ++c) {
    float xv = xb[(size_t)c * L];
    aq = fmaf(wqr[c], xv, aq);
    ak = fmaf(wkr[c], xv, ak);
    av = fmaf(wvr[c], xv, av);
  }
  q16[((size_t)b * L + l) * C + o]  = (_Float16)aq;
  k16[((size_t)b * L + l) * C + o]  = (_Float16)ak;
  v16t[((size_t)b * C + o) * L + l] = (_Float16)av;
}

// ---------------------------------------------------------------------------
// Kernel 2: flash attention + residual.
// 128 threads (4 waves) per block; 64 query rows per block, 16 per wave.
// K/V tiles double-buffered in LDS via async global->LDS loads.
// ---------------------------------------------------------------------------
__global__ __launch_bounds__(128) void attn_fwd(
    const float* __restrict__ x,
    const _Float16* __restrict__ q16, const _Float16* __restrict__ k16,
    const _Float16* __restrict__ v16t, float* __restrict__ out) {
  __shared__ __attribute__((aligned(16))) _Float16 sQ[64 * LDSS];
  __shared__ __attribute__((aligned(16))) _Float16 sK[2][64 * LDSS];
  __shared__ __attribute__((aligned(16))) _Float16 sV[2][64 * LDSS];
  __shared__ __attribute__((aligned(16))) _Float16 sP[4 * 16 * LDSS];

  const int tid  = threadIdx.x;
  const int lane = tid & 31;
  const int wid  = tid >> 5;
  const int nh   = lane & 15;
  const int hi   = lane >> 4;
  const int tiles = L / 64;
  const int b  = blockIdx.x / tiles;
  const int i0 = (blockIdx.x % tiles) * 64;

  // --- stage Q tile (64 rows x 64 f16) ---
  {
    const unsigned int* qg =
        reinterpret_cast<const unsigned int*>(q16 + ((size_t)b * L + i0) * C);
    unsigned int* sq = reinterpret_cast<unsigned int*>(sQ);
    for (int t = tid; t < 64 * 32; t += 128) {
      int r = t >> 5, c0 = t & 31;
      sq[r * (LDSS / 2) + c0] = qg[r * 32 + c0];
    }
  }
  __syncthreads();

  const int qm = wid * 16 + nh;
  v16h qa0 = load_A(sQ, qm, hi, 0);
  v16h qa1 = load_A(sQ, qm, hi, 32);

  float mrow[8], lrow[8];
  const v8f vzero = {0.f, 0.f, 0.f, 0.f, 0.f, 0.f, 0.f, 0.f};
  v8f oacc[4];
#pragma unroll
  for (int r = 0; r < 8; ++r) { mrow[r] = -__builtin_inff(); lrow[r] = 0.f; }
#pragma unroll
  for (int sN = 0; sN < 4; ++sN) oacc[sN] = vzero;

  // global tile bases (byte pointers)
  const char* kgb0 = (const char*)k16 + (size_t)b * L * (C * 2);      // +j0*128
  const char* vgb0 = (const char*)v16t + (size_t)b * C * L * 2;       // +j0*2
  const unsigned sKb[2] = {(unsigned)(uintptr_t)&sK[0][0],
                           (unsigned)(uintptr_t)&sK[1][0]};
  const unsigned sVb[2] = {(unsigned)(uintptr_t)&sV[0][0],
                           (unsigned)(uintptr_t)&sV[1][0]};
  _Float16* sPw = sP + wid * 16 * LDSS;

  // per-thread async stage of one 64-key K/V tile: 8x b128 (4 K + 4 V)
  auto issue_tile = [&](int buf, int j0) {
#pragma unroll
    for (int n = 0; n < 4; ++n) {
      int t = tid + n * 128;           // 0..511
      int r = t >> 3, c16 = t & 7;     // row, 16B column
      async_ld_b128(sKb[buf] + r * (LDSS * 2) + c16 * 16,
                    kgb0 + ((size_t)(j0 + r)) * 128 + c16 * 16);
      async_ld_b128(sVb[buf] + r * (LDSS * 2) + c16 * 16,
                    vgb0 + ((size_t)r * L + j0) * 2 + c16 * 16);
    }
  };

  issue_tile(0, 0);
  int cur = 0;

  for (int j0 = 0; j0 < L; j0 += 64) {
    const bool has_next = (j0 + 64) < L;
    if (has_next) issue_tile(cur ^ 1, j0 + 64);   // overlap with compute
    // async ops complete in order per wave: <=8 in flight -> current tile done
    if (has_next) wait_asynccnt<8>(); else wait_asynccnt<0>();
    __syncthreads();   // all threads' loads for `cur` complete & visible

    const _Float16* sKc = sK[cur];
    const _Float16* sVc = sV[cur];

    // S = Q * K^T (four 16x16 j-subtiles, K-dim = 64 channels)
    v8f st[4];
#pragma unroll
    for (int jt = 0; jt < 4; ++jt) {
      v16h bk0 = load_B(sKc, jt * 16 + nh, hi, 0);
      v16h bk1 = load_B(sKc, jt * 16 + nh, hi, 32);
      v8f acc = wmma_f16(qa0, bk0, vzero);
      acc     = wmma_f16(qa1, bk1, acc);
      st[jt]  = acc * 0.125f;   // 1/sqrt(C)
    }

    // online softmax update (row stats across 16 lanes of each half)
#pragma unroll
    for (int r = 0; r < 8; ++r) {
      float m = fmaxf(fmaxf(st[0][r], st[1][r]), fmaxf(st[2][r], st[3][r]));
#pragma unroll
      for (int off = 1; off < 16; off <<= 1)
        m = fmaxf(m, __shfl_xor(m, off, 32));
      float mn   = fmaxf(mrow[r], m);
      float corr = __expf(mrow[r] - mn);
      mrow[r] = mn;
      float s = 0.f;
#pragma unroll
      for (int jt = 0; jt < 4; ++jt) {
        float p = __expf(st[jt][r] - mn);
        st[jt][r] = p;
        s += p;
      }
#pragma unroll
      for (int off = 1; off < 16; off <<= 1)
        s += __shfl_xor(s, off, 32);
      lrow[r] = lrow[r] * corr + s;
#pragma unroll
      for (int sN = 0; sN < 4; ++sN) oacc[sN][r] *= corr;
    }

    // D-layout -> A-layout conversion via per-wave LDS scratch
#pragma unroll
    for (int jt = 0; jt < 4; ++jt)
#pragma unroll
      for (int r = 0; r < 8; ++r)
        sPw[(r + 8 * hi) * LDSS + jt * 16 + nh] = (_Float16)st[jt][r];
    asm volatile("" ::: "memory");  // DS ops are in-order within a wave

    v16h pa0 = load_A(sPw, nh, hi, 0);
    v16h pa1 = load_A(sPw, nh, hi, 32);

    // O += P * V (K-dim = 64 keys of this step; four 16-col c-subtiles)
#pragma unroll
    for (int sN = 0; sN < 4; ++sN) {
      v16h bv0 = load_B(sVc, sN * 16 + nh, hi, 0);
      v16h bv1 = load_B(sVc, sN * 16 + nh, hi, 32);
      oacc[sN] = wmma_f16(pa0, bv0, oacc[sN]);
      oacc[sN] = wmma_f16(pa1, bv1, oacc[sN]);
    }

    __syncthreads();   // everyone done reading `cur` before it is refilled
    cur ^= 1;
  }

  // epilogue: normalize, add residual, write [B,C,L]
#pragma unroll
  for (int sN = 0; sN < 4; ++sN) {
    int c = sN * 16 + nh;
#pragma unroll
    for (int r = 0; r < 8; ++r) {
      int l = i0 + wid * 16 + r + 8 * hi;
      size_t idx = ((size_t)(b * C + c)) * L + l;
      out[idx] = oacc[sN][r] / lrow[r] + x[idx];
    }
  }
}

// ---------------------------------------------------------------------------
extern "C" void kernel_launch(void* const* d_in, const int* in_sizes, int n_in,
                              void* d_out, int out_size, void* d_ws,
                              size_t ws_size, hipStream_t stream) {
  const float* x  = (const float*)d_in[0];
  const float* wq = (const float*)d_in[1];
  const float* bq = (const float*)d_in[2];
  const float* wk = (const float*)d_in[3];
  const float* bk = (const float*)d_in[4];
  const float* wv = (const float*)d_in[5];
  const float* bv = (const float*)d_in[6];
  float* out = (float*)d_out;

  _Float16* q16  = (_Float16*)d_ws;
  _Float16* k16  = q16 + (size_t)Bn * L * C;
  _Float16* v16t = k16 + (size_t)Bn * L * C;

  proj_qkv<<<dim3(Bn * C * (L / 256)), dim3(256), 0, stream>>>(
      x, wq, bq, wk, bk, wv, bv, q16, k16, v16t);
  attn_fwd<<<dim3(Bn * (L / 64)), dim3(128), 0, stream>>>(
      x, q16, k16, v16t, out);
}
